// SelfAttention_77756087927441
// MI455X (gfx1250) — matrix operand order repaired and checked by reading, not compiled, not run on hardware
//
#include <hip/hip_runtime.h>
#include <hip/hip_bf16.h>

// ---------------------------------------------------------------------------
// Self-attention, single "head" of width 768.
//   B=8, S=2048, E=768.  fp32 in / fp32 out, bf16 WMMA internally.
//   Projections via WMMA NT-GEMM; attention via single-pass flash with the
//   full 16x768 output tile register-resident; K/V/Q tiles staged into LDS
//   by the Tensor Data Mover, software-pipelined on TENSORcnt partial waits.
// ---------------------------------------------------------------------------

typedef __bf16 bf16_t;
typedef __attribute__((ext_vector_type(16))) __bf16 v16bf;
typedef __attribute__((ext_vector_type(8)))  __bf16 v8bf;
typedef __attribute__((ext_vector_type(4)))  __bf16 v4bf;
typedef __attribute__((ext_vector_type(8)))  float  v8f;
typedef __attribute__((ext_vector_type(4)))  unsigned int v4u;
typedef __attribute__((ext_vector_type(8)))  int    v8i;
typedef __attribute__((ext_vector_type(4)))  int    v4i;

static constexpr int BATCH = 8;
static constexpr int SEQ   = 2048;
static constexpr int EMB   = 768;
static constexpr int NKT   = SEQ / 32;               // 64 key tiles
static constexpr float SM_SCALE = 0.036084391824351615f; // 1/sqrt(768)
static constexpr float LOG2E    = 1.4426950408889634f;

// --- WMMA helper: D = A(16x32 bf16) * B(32x16 bf16) + C(16x16 f32) ----------
__device__ __forceinline__ v8f wmma_bf16(v16bf a, v16bf b, v8f c) {
    return __builtin_amdgcn_wmma_f32_16x16x32_bf16(
        /*neg_a=*/false, a, /*neg_b=*/false, b,
        /*c_mod=*/(short)0, c, /*reuse_a=*/false, /*reuse_b=*/false);
}

// load 16 contiguous bf16 (order used here for both A and B: per-lane contiguous 16 K values)
__device__ __forceinline__ v16bf ld16(const bf16_t* p) {
    v8bf lo = *(const v8bf*)p;
    v8bf hi = *(const v8bf*)(p + 8);
    v16bf r;
#pragma unroll
    for (int i = 0; i < 8; ++i) { r[i] = lo[i]; r[i + 8] = hi[i]; }
    return r;
}

// A-operand layout: per-lane two 8-element chunks 16 apart (K {o..o+7, o+16..o+23})
__device__ __forceinline__ v16bf ld16_split(const bf16_t* p) {
    v8bf lo = *(const v8bf*)p;
    v8bf hi = *(const v8bf*)(p + 16);
    v16bf r;
#pragma unroll
    for (int i = 0; i < 8; ++i) { r[i] = lo[i]; r[i + 8] = hi[i]; }
    return r;
}

// ---------------------------------------------------------------------------
// Tensor Data Mover: issue a 2D bf16 tile load  global -> LDS.
//   rowlen  : elements per row (contiguous in memory and in LDS)
//   nrows   : number of rows
//   rstride : row stride in global memory, in elements
// D# layout per CDNA5 ISA ch.8 (group0 128b, group1 256b; groups 2/3 = 0 for 2D).
// ---------------------------------------------------------------------------
#if __has_builtin(__builtin_amdgcn_tensor_load_to_lds)
#define USE_TDM 1
#else
#define USE_TDM 0
#endif

__device__ __forceinline__ void tdm_load_2d_bf16(unsigned lds_off, const void* gptr,
                                                 unsigned rowlen, unsigned nrows,
                                                 unsigned long long rstride) {
#if USE_TDM
    unsigned long long ga = (unsigned long long)gptr;
    unsigned long long sb = rstride;               // dim0 stride, elements
    v4u g0;
    g0[0] = 1u;                                    // count=1, user descriptor
    g0[1] = lds_off;                               // LDS byte address
    g0[2] = (unsigned)(ga & 0xffffffffu);
    g0[3] = (unsigned)((ga >> 32) & 0x01ffffffu) | (2u << 30);  // type=2 ("image")
    v8i g1;
    g1[0] = (int)(1u << 16);                       // data_size=1 (2 bytes), wg_mask=0
    g1[1] = (int)((rowlen & 0xffffu) << 16);       // tensor_dim0[15:0]  @bit48
    g1[2] = (int)(((rowlen >> 16) & 0xffffu) | ((nrows & 0xffffu) << 16)); // td0 hi | td1 lo
    g1[3] = (int)(((nrows >> 16) & 0xffffu) | ((rowlen & 0xffffu) << 16)); // td1 hi | tile_dim0
    g1[4] = (int)(nrows & 0xffffu);                // tile_dim1 | tile_dim2=0
    g1[5] = (int)(sb & 0xffffffffu);               // tensor_dim0_stride[31:0]
    g1[6] = (int)((sb >> 32) & 0xffffu);           // stride hi | tensor_dim1_stride lo = 0
    g1[7] = 0;
    v4i gz = {0, 0, 0, 0};
#if __has_include(<hip/amd_detail/amd_gfx1250_TDM.h>)
    v8i gz8 = {0, 0, 0, 0, 0, 0, 0, 0};
    __builtin_amdgcn_tensor_load_to_lds(g0, g1, gz, gz, gz8, 0);   // clang-23 form
#else
    __builtin_amdgcn_tensor_load_to_lds(g0, g1, gz, gz, 0);        // ROCm 7.2 form
#endif
#endif
}

// ---------------------------------------------------------------------------
// Kernel 1: fp32 -> bf16 conversion (vectorized x4)
// ---------------------------------------------------------------------------
__global__ __launch_bounds__(256)
void cvt_f32_bf16(const float* __restrict__ in, bf16_t* __restrict__ out, int n4) {
    int i = blockIdx.x * 256 + threadIdx.x;
    if (i >= n4) return;
    float4 v = reinterpret_cast<const float4*>(in)[i];
    v4bf o = { (bf16_t)v.x, (bf16_t)v.y, (bf16_t)v.z, (bf16_t)v.w };
    reinterpret_cast<v4bf*>(out)[i] = o;
}

// ---------------------------------------------------------------------------
// Kernel 2: NT-GEMM  C[m][n] = sum_k A[m][k]*B[n][k]   (bf16 in, bf16 out)
// Block tile 128x128, 8 waves (4 along M x 2 along N), wave tile 32x64.
// ---------------------------------------------------------------------------
__global__ __launch_bounds__(256, 2)
void gemm_nt_bf16(const bf16_t* __restrict__ A, size_t sAb,
                  const bf16_t* __restrict__ B, size_t sBb,
                  bf16_t* __restrict__ C, size_t sCb,
                  int M, int N, int K) {
    const int tid  = threadIdx.x;
    const int wid  = tid >> 5;
    const int lane = tid & 31;
    const int l16  = lane & 15;
    const int hh   = lane >> 4;          // lane half (0 or 1)
    const int wm   = (wid & 3) * 32;     // wave offset in M within block
    const int wn   = (wid >> 2) * 64;    // wave offset in N within block

    const bf16_t* Ab = A + (size_t)blockIdx.z * sAb +
                       (size_t)(blockIdx.x * 128 + wm) * K;
    const bf16_t* Bb = B + (size_t)blockIdx.z * sBb +
                       (size_t)(blockIdx.y * 128 + wn) * K;

    v8f acc[2][4];
#pragma unroll
    for (int i = 0; i < 2; ++i)
#pragma unroll
        for (int j = 0; j < 4; ++j) acc[i][j] = {};

    const int ksteps = K >> 5;
    for (int c = 0; c < ksteps; ++c) {
        const int ko = c * 32;
        v16bf a[2], bb[4];
#pragma unroll
        for (int i = 0; i < 2; ++i)
            a[i] = ld16(Ab + (size_t)(i * 16 + l16) * K + ko + hh * 16);
#pragma unroll
        for (int j = 0; j < 4; ++j)
            bb[j] = ld16(Bb + (size_t)(j * 16 + l16) * K + ko + hh * 16);
#pragma unroll
        for (int i = 0; i < 2; ++i)
#pragma unroll
            for (int j = 0; j < 4; ++j)
                acc[i][j] = wmma_bf16(a[i], bb[j], acc[i][j]);
    }

    bf16_t* Cb = C + (size_t)blockIdx.z * sCb;
#pragma unroll
    for (int i = 0; i < 2; ++i)
#pragma unroll
        for (int j = 0; j < 4; ++j)
#pragma unroll
            for (int r = 0; r < 8; ++r) {
                int row = blockIdx.x * 128 + wm + i * 16 + r + hh * 8;
                int col = blockIdx.y * 128 + wn + j * 16 + l16;
                Cb[(size_t)row * N + col] = (bf16_t)acc[i][j][r];
            }
}

// ---------------------------------------------------------------------------
// Kernel 3: single-pass flash attention, TDM-pipelined.
//   Block = 256 threads (8 waves), 128 query rows of one batch per block.
//   Each wave: 16 q-rows x full 768 output dims (acc = 48 x v8f = 384 VGPRs).
//   LDS: Q block 192KB + K tile 48KB + V^T tile 48KB + P scratch 8KB = 296KB.
//   Pipeline (issue order ... K[t], V[t], K[t+1], V[t+1] ... is TENSORcnt
//   in-order, so partial waits gate exactly the tile we need):
//     wait tensorcnt<=1  -> K[t] ready     | QK^T        (V[t] DMA in flight)
//     barrier, issue K[t+1]                | softmax     (K[t+1] DMA in flight)
//     wait tensorcnt<=1* -> V[t] ready     | P·V
//     barrier, issue V[t+1]                | next QK^T   (V[t+1] DMA in flight)
//   (* <=0 on the last tile, where K[t+1] was not issued)
// ---------------------------------------------------------------------------
__global__ __launch_bounds__(256, 1)
void flash_attn(const bf16_t* __restrict__ Q, const bf16_t* __restrict__ K,
                const bf16_t* __restrict__ Vt, const int* __restrict__ mask,
                float* __restrict__ out) {
    __shared__ __align__(16) bf16_t ldsQ[128 * EMB];   // [128][768]
    __shared__ __align__(16) bf16_t ldsK[32 * EMB];    // [32][768]
    __shared__ __align__(16) bf16_t ldsV[EMB * 32];    // [768][32]  (V^T tile)
    __shared__ __align__(16) bf16_t ldsP[8 * 16 * 32]; // per-wave 16x32 probs

    const int b      = blockIdx.z;
    const int qblock = blockIdx.x * 128;
    const int tid    = threadIdx.x;
    const int wid    = tid >> 5;
    const int lane   = tid & 31;
    const int l16    = lane & 15;
    const int hh     = lane >> 4;
    const int qr0    = wid * 16;           // wave's q-row offset in block

    const bf16_t* Qb = Q  + (size_t)b * SEQ * EMB;
    const bf16_t* Kb = K  + (size_t)b * SEQ * EMB;
    const bf16_t* Vb = Vt + (size_t)b * EMB * SEQ;
    const int*    Mb = mask + (size_t)b * SEQ * SEQ;

    const unsigned offQ = (unsigned)(uintptr_t)&ldsQ[0];
    const unsigned offK = (unsigned)(uintptr_t)&ldsK[0];
    const unsigned offV = (unsigned)(uintptr_t)&ldsV[0];

#if USE_TDM
    // Prefetch Q block + first K/V tiles (in-order: Q, K0, V0).
    if (wid == 0) {
        tdm_load_2d_bf16(offQ, Qb + (size_t)qblock * EMB, EMB, 128, EMB);
        tdm_load_2d_bf16(offK, Kb, EMB, 32, EMB);
        tdm_load_2d_bf16(offV, Vb, 32, EMB, SEQ);
    }
#else
    {
        const uint4* src = (const uint4*)(Qb + (size_t)qblock * EMB);
        uint4* dst = (uint4*)ldsQ;
        for (int i = tid; i < (128 * EMB) / 8; i += 256) dst[i] = src[i];
    }
#endif

    v8f acc[48];
#pragma unroll
    for (int dt = 0; dt < 48; ++dt) acc[dt] = {};
    float m_i[8], l_i[8];
#pragma unroll
    for (int r = 0; r < 8; ++r) { m_i[r] = -3.0e38f; l_i[r] = 0.0f; }

    bf16_t* Pw = ldsP + wid * (16 * 32);
    const int qrow = qblock + qr0;

    for (int kt = 0; kt < NKT; ++kt) {
        const int kbase = kt * 32;
#if USE_TDM
        // K[kt] (and everything issued before it) is in LDS; V[kt] may still fly.
        if (wid == 0) __builtin_amdgcn_s_wait_tensorcnt((short)1);
        __syncthreads();
#else
        __syncthreads();
        {
            const uint4* src = (const uint4*)(Kb + (size_t)kbase * EMB);
            uint4* dst = (uint4*)ldsK;
            for (int i = tid; i < (32 * EMB) / 8; i += 256) dst[i] = src[i];
        }
        {
            uint4* dst = (uint4*)ldsV;
            for (int i = tid; i < (EMB * 32) / 8; i += 256) {
                int r = i >> 2, pc = i & 3;
                dst[i] = *(const uint4*)(Vb + (size_t)r * SEQ + kbase + pc * 8);
            }
        }
        __syncthreads();
#endif

        // ---- S tile: Q(16x768) x K_tile^T -> two 16x16 f32 tiles ----
        v8f s0 = {}, s1 = {};
#pragma unroll
        for (int c = 0; c < EMB / 32; ++c) {
            v16bf a  = ld16(&ldsQ[(size_t)(qr0 + l16) * EMB + c * 32 + hh * 16]);
            v16bf b0 = ld16(&ldsK[(size_t)l16 * EMB + c * 32 + hh * 16]);
            v16bf b1 = ld16(&ldsK[(size_t)(l16 + 16) * EMB + c * 32 + hh * 16]);
            s0 = wmma_bf16(a, b0, s0);
            s1 = wmma_bf16(a, b1, s1);
        }

#if USE_TDM
        // ldsK consumed by all waves -> overlap next K tile DMA with softmax+PV.
        __syncthreads();
        if (wid == 0 && kt + 1 < NKT)
            tdm_load_2d_bf16(offK, Kb + (size_t)(kbase + 32) * EMB, EMB, 32, EMB);
#endif

        // ---- mask + scale ----
        float z0[8], z1[8];
#pragma unroll
        for (int r = 0; r < 8; ++r) {
            int m = r + hh * 8;
            const int* mp = Mb + (size_t)(qrow + m) * SEQ + kbase;
            float v0 = (mp[l16]      == 0) ? -1.0e20f : s0[r];
            float v1 = (mp[l16 + 16] == 0) ? -1.0e20f : s1[r];
            z0[r] = v0 * SM_SCALE;
            z1[r] = v1 * SM_SCALE;
        }

        // ---- online softmax: row max over the 32 keys of this tile ----
        float tmax[8];
#pragma unroll
        for (int r = 0; r < 8; ++r) tmax[r] = fmaxf(z0[r], z1[r]);
#pragma unroll
        for (int off = 1; off < 16; off <<= 1)
#pragma unroll
            for (int r = 0; r < 8; ++r)
                tmax[r] = fmaxf(tmax[r], __shfl_xor(tmax[r], off, 32));

        float mnew[8], p0[8], p1[8], tsum[8], corr[8];
        bool changed = false;
#pragma unroll
        for (int r = 0; r < 8; ++r) {
            mnew[r] = fmaxf(m_i[r], tmax[r]);
            p0[r] = __builtin_amdgcn_exp2f((z0[r] - mnew[r]) * LOG2E);
            p1[r] = __builtin_amdgcn_exp2f((z1[r] - mnew[r]) * LOG2E);
            tsum[r] = p0[r] + p1[r];
            corr[r] = __builtin_amdgcn_exp2f((m_i[r] - mnew[r]) * LOG2E);
            changed |= (corr[r] != 1.0f);
            m_i[r] = mnew[r];
        }
#pragma unroll
        for (int off = 1; off < 16; off <<= 1)
#pragma unroll
            for (int r = 0; r < 8; ++r)
                tsum[r] += __shfl_xor(tsum[r], off, 32);
#pragma unroll
        for (int r = 0; r < 8; ++r) l_i[r] = l_i[r] * corr[r] + tsum[r];

        // ---- rescale accumulator (only when a row max actually moved) ----
        if (__ballot(changed)) {
#pragma unroll
            for (int dt = 0; dt < 48; ++dt)
#pragma unroll
                for (int r = 0; r < 8; ++r) acc[dt][r] *= corr[r];
        }

        // ---- C-layout probs -> LDS -> A-layout fragment (bf16) ----
#pragma unroll
        for (int r = 0; r < 8; ++r) {
            int m = r + hh * 8;
            Pw[m * 32 + l16]      = (bf16_t)p0[r];
            Pw[m * 32 + 16 + l16] = (bf16_t)p1[r];
        }
        asm volatile("s_wait_dscnt 0x0" ::: "memory");  // wave-local LDS RAW
        v16bf aP = ld16(&Pw[l16 * 32 + hh * 16]);

#if USE_TDM
        // Gate on V[kt]: outstanding = {V[kt], K[kt+1]} except on the last tile.
        if (wid == 0) {
            if (kt + 1 < NKT) __builtin_amdgcn_s_wait_tensorcnt((short)1);
            else              __builtin_amdgcn_s_wait_tensorcnt((short)0);
        }
        __syncthreads();
#endif

        // ---- acc += P(16x32) x V_tile(32x16 per dt) ----
#pragma unroll
        for (int dt = 0; dt < 48; ++dt) {
            v16bf bV = ld16(&ldsV[(size_t)(dt * 16 + l16) * 32 + hh * 16]);
            acc[dt] = wmma_bf16(aP, bV, acc[dt]);
        }

#if USE_TDM
        // ldsV consumed -> overlap next V tile DMA with next iteration's QK^T.
        __syncthreads();
        if (wid == 0 && kt + 1 < NKT)
            tdm_load_2d_bf16(offV, Vb + (kbase + 32), 32, EMB, SEQ);
#endif
    }

    // ---- normalize and write fp32 output ----
    float* Ob = out + (size_t)b * SEQ * EMB;
#pragma unroll
    for (int r = 0; r < 8; ++r) {
        float inv = 1.0f / l_i[r];
        int m = r + hh * 8;
        float* orow = Ob + (size_t)(qrow + m) * EMB;
#pragma unroll
        for (int dt = 0; dt < 48; ++dt)
            orow[dt * 16 + l16] = acc[dt][r] * inv;
    }
}

// ---------------------------------------------------------------------------
// Host-side launcher
// ---------------------------------------------------------------------------
extern "C" void kernel_launch(void* const* d_in, const int* in_sizes, int n_in,
                              void* d_out, int out_size, void* d_ws, size_t ws_size,
                              hipStream_t stream) {
    // setup_inputs order: x, Wv, Wk, Wq, mask
    const float* x    = (const float*)d_in[0];
    const float* Wv   = (const float*)d_in[1];
    const float* Wk   = (const float*)d_in[2];
    const float* Wq   = (const float*)d_in[3];
    const int*   mask = (const int*)d_in[4];
    float* out = (float*)d_out;

    const size_t SE = (size_t)SEQ * EMB;          // per-batch x/Q/K elements
    const size_t nX = (size_t)BATCH * SE;         // 12.58M
    const size_t nW = (size_t)EMB * EMB;          // 589824

    char* ws = (char*)d_ws;
    bf16_t* xb  = (bf16_t*)ws;            ws += nX * sizeof(bf16_t);
    bf16_t* Wqb = (bf16_t*)ws;            ws += nW * sizeof(bf16_t);
    bf16_t* Wkb = (bf16_t*)ws;            ws += nW * sizeof(bf16_t);
    bf16_t* Wvb = (bf16_t*)ws;            ws += nW * sizeof(bf16_t);
    bf16_t* Qb  = (bf16_t*)ws;            ws += nX * sizeof(bf16_t);
    bf16_t* Kb  = (bf16_t*)ws;            ws += nX * sizeof(bf16_t);
    bf16_t* Vtb = (bf16_t*)ws;            ws += nX * sizeof(bf16_t);   // [B][768][2048]

    // 1) fp32 -> bf16
    {
        int n4 = (int)(nX / 4);
        cvt_f32_bf16<<<(n4 + 255) / 256, 256, 0, stream>>>(x, xb, n4);
        int w4 = (int)(nW / 4);
        cvt_f32_bf16<<<(w4 + 255) / 256, 256, 0, stream>>>(Wq, Wqb, w4);
        cvt_f32_bf16<<<(w4 + 255) / 256, 256, 0, stream>>>(Wk, Wkb, w4);
        cvt_f32_bf16<<<(w4 + 255) / 256, 256, 0, stream>>>(Wv, Wvb, w4);
    }

    // 2) projections (NT-GEMM, bf16)
    //    Q = x (2048x768) * Wq^T            -> [B][2048][768]
    //    K = x * Wk^T                        -> [B][2048][768]
    //    V^T = Wv (768x768) * x^T            -> [B][768][2048]
    gemm_nt_bf16<<<dim3(SEQ / 128, EMB / 128, BATCH), 256, 0, stream>>>(
        xb, SE, Wqb, 0, Qb, SE, SEQ, EMB, EMB);
    gemm_nt_bf16<<<dim3(SEQ / 128, EMB / 128, BATCH), 256, 0, stream>>>(
        xb, SE, Wkb, 0, Kb, SE, SEQ, EMB, EMB);
    gemm_nt_bf16<<<dim3(EMB / 128, SEQ / 128, BATCH), 256, 0, stream>>>(
        Wvb, 0, xb, SE, Vtb, SE, EMB, SEQ, EMB);

    // 3) single-pass flash attention (fp32 out)
    flash_attn<<<dim3(SEQ / 128, 1, BATCH), 256, 0, stream>>>(
        Qb, Kb, Vtb, mask, out);
}
